// MultiHeadAttention_79345225826439
// MI455X (gfx1250) — matrix operand order repaired and checked by reading, not compiled, not run on hardware
//
#include <hip/hip_runtime.h>
#include <hip/hip_bf16.h>

// ---------------------------------------------------------------------------
// MI455X (gfx1250) multi-head attention forward.
//   Stage 0: fp32 -> bf16 conversion of activations + weights (workspace).
//   Stage 1: Q/K/V projections via v_wmma_f32_16x16x32_bf16.
//            16x64 tile per wave (4 accumulators share one A fragment) with
//            explicit ping-pong double buffering so global_load_b128 of the
//            next K-step overlaps the current WMMAs (partial loadcnt waits).
//   Stage 2: per-16-query-stripe: scores (WMMA, double-buffered B frags)
//            -> LDS, masked, wave32 shuffle softmax, fp32 attn_weights
//            streamed to HBM once, bf16 copy kept in LDS and fed back into
//            WMMA for out = P @ V (K split across all 8 waves + LDS reduce).
// ---------------------------------------------------------------------------

typedef __attribute__((ext_vector_type(16))) __bf16 v16bf;
typedef __attribute__((ext_vector_type(8)))  float  v8f;

#define D_MODEL   1024
#define NUM_HEADS 16
#define D_K       64
#define BATCH     2
#define SEQ       2048
#define M_ROWS    (BATCH * SEQ)          // 4096
#define NEG_BIG   (-1.0e9f)

union FragAB {
    uint4  u4[2];
    v16bf  v;
};

__device__ __forceinline__ unsigned short f32_to_bf16_rne(float f) {
    unsigned int x = __float_as_uint(f);
    unsigned int r = x + 0x7FFFu + ((x >> 16) & 1u);   // round-to-nearest-even
    return (unsigned short)(r >> 16);
}

#define WMMA_BF16(A, B, C) \
    __builtin_amdgcn_wmma_f32_16x16x32_bf16(false, (A).v, false, (B).v, \
                                            (short)0, (C), false, false)

// ---------------------------------------------------------------- stage 0 ---
__global__ __launch_bounds__(256)
void cvt_f32_to_bf16(const float* __restrict__ src,
                     unsigned short* __restrict__ dst, int n) {
    int i = blockIdx.x * blockDim.x + threadIdx.x;
    int stride = gridDim.x * blockDim.x;
    for (; i < n; i += stride) dst[i] = f32_to_bf16_rne(src[i]);
}

// ---------------------------------------------------------------- stage 1 ---
// C[m,n] = sum_k X[m,k] * W[n,k] + bias[n]   (nn.Linear: x @ W^T + b)
// One 16x64 output tile per wave; explicit double-buffered K loop.
__global__ __launch_bounds__(256)
void proj_gemm_bf16(const unsigned short* __restrict__ X,
                    const unsigned short* __restrict__ W,
                    const float* __restrict__ bias,
                    unsigned short* __restrict__ dst,
                    int transposed)
{
    const int lane = threadIdx.x & 31;
    const int wave = threadIdx.x >> 5;
    const int tile = blockIdx.x * 8 + wave;   // 512 blocks * 8 waves = 4096 tiles
    const int m0 = (tile >> 4) << 4;          // 256 row tiles (16 rows)
    const int n0 = (tile & 15) << 6;          // 16 col supertiles (64 cols)
    const int g  = lane >> 4;
    const int ln = lane & 15;

    const unsigned short* xrow = X + (size_t)(m0 + ln) * D_MODEL + 16 * g;  // A row M=ln
    const unsigned short* wr0  = W + (size_t)(n0 +  0 + ln) * D_MODEL + 16 * g;
    const unsigned short* wr1  = W + (size_t)(n0 + 16 + ln) * D_MODEL + 16 * g;
    const unsigned short* wr2  = W + (size_t)(n0 + 32 + ln) * D_MODEL + 16 * g;
    const unsigned short* wr3  = W + (size_t)(n0 + 48 + ln) * D_MODEL + 16 * g;

#define PROJ_LOAD(a_, b0_, b1_, b2_, b3_, k_)                                 \
    do {                                                                      \
        (a_).u4[0]  = *(const uint4*)(xrow + (k_));                           \
        (a_).u4[1]  = *(const uint4*)(xrow + (k_) + 8);                       \
        (b0_).u4[0] = *(const uint4*)(wr0 + (k_));                            \
        (b0_).u4[1] = *(const uint4*)(wr0 + (k_) + 8);                        \
        (b1_).u4[0] = *(const uint4*)(wr1 + (k_));                            \
        (b1_).u4[1] = *(const uint4*)(wr1 + (k_) + 8);                        \
        (b2_).u4[0] = *(const uint4*)(wr2 + (k_));                            \
        (b2_).u4[1] = *(const uint4*)(wr2 + (k_) + 8);                        \
        (b3_).u4[0] = *(const uint4*)(wr3 + (k_));                            \
        (b3_).u4[1] = *(const uint4*)(wr3 + (k_) + 8);                        \
    } while (0)

    v8f ac0 = {}, ac1 = {}, ac2 = {}, ac3 = {};
    FragAB aX, b0X, b1X, b2X, b3X;     // ping
    FragAB aY, b0Y, b1Y, b2Y, b3Y;     // pong
    PROJ_LOAD(aX, b0X, b1X, b2X, b3X, 0);

    for (int k0 = 0; k0 < D_MODEL; k0 += 64) {
        __builtin_prefetch(xrow + k0 + 256, 0, 1);         // global_prefetch_b8
        PROJ_LOAD(aY, b0Y, b1Y, b2Y, b3Y, k0 + 32);        // overlap with X WMMAs
        ac0 = WMMA_BF16(aX, b0X, ac0);
        ac1 = WMMA_BF16(aX, b1X, ac1);
        ac2 = WMMA_BF16(aX, b2X, ac2);
        ac3 = WMMA_BF16(aX, b3X, ac3);
        if (k0 + 64 < D_MODEL)
            PROJ_LOAD(aX, b0X, b1X, b2X, b3X, k0 + 64);    // overlap with Y WMMAs
        ac0 = WMMA_BF16(aY, b0Y, ac0);
        ac1 = WMMA_BF16(aY, b1Y, ac1);
        ac2 = WMMA_BF16(aY, b2Y, ac2);
        ac3 = WMMA_BF16(aY, b3Y, ac3);
    }
#undef PROJ_LOAD

    v8f* accs[4] = { &ac0, &ac1, &ac2, &ac3 };
    for (int j = 0; j < 4; ++j) {
        const int n = n0 + 16 * j + ln;          // C: N = ln (fixed per lane)
        const float bv = bias[n];
        v8f acc = *accs[j];
        if (!transposed) {
            for (int r = 0; r < 8; ++r) {
                int m = m0 + r + 8 * g;          // C: M = r + 8*(lane/16)
                dst[(size_t)m * D_MODEL + n] = f32_to_bf16_rne(acc[r] + bv);
            }
        } else {
            for (int r = 0; r < 8; ++r) {
                int m = m0 + r + 8 * g;
                int b_ = m >> 11, s = m & (SEQ - 1);
                dst[(size_t)b_ * (D_MODEL * SEQ) + (size_t)n * SEQ + s] =
                    f32_to_bf16_rne(acc[r] + bv);
            }
        }
    }
}

// ---------------------------------------------------------------- stage 2 ---
// One workgroup = one (b, h, 16-query stripe).  8 waves.
// LDS: 16x2048 fp32 scores (128 KB) + 16x2048 bf16 probs (64 KB) = 192 KB.
__global__ __launch_bounds__(256)
void attn_fused(const unsigned short* __restrict__ Q,   // [4096,1024] bf16
                const unsigned short* __restrict__ K,   // [4096,1024] bf16
                const unsigned short* __restrict__ Vt,  // [B,1024,2048] bf16
                const int*            __restrict__ mask,// [B,2048,2048]
                float* __restrict__ out,                // [B,2048,1024]
                float* __restrict__ attnw)              // [B,16,2048,2048]
{
    extern __shared__ char smem[];
    float*          sc = (float*)smem;                            // [16][2048]
    unsigned short* pa = (unsigned short*)(smem + 16 * SEQ * 4);  // [16][2048] bf16

    const int lane = threadIdx.x & 31;
    const int wave = threadIdx.x >> 5;
    const int g  = lane >> 4;
    const int ln = lane & 15;

    const int qt = blockIdx.x & 127;           // S/16 = 128 stripes
    const int h  = (blockIdx.x >> 7) & 15;
    const int b  = blockIdx.x >> 11;
    const int q0 = qt << 4;

    // ---- scores = (Q K^T) * scale, masked, into LDS --------------------
    const unsigned short* qrow =
        Q + ((size_t)(b * SEQ + q0 + ln)) * D_MODEL + h * D_K;
    FragAB a0, a1;                             // A-frags for d0 = 0 and 32
    a0.u4[0] = *(const uint4*)(qrow + 0  + 16 * g);
    a0.u4[1] = *(const uint4*)(qrow + 8  + 16 * g);
    a1.u4[0] = *(const uint4*)(qrow + 32 + 16 * g);
    a1.u4[1] = *(const uint4*)(qrow + 40 + 16 * g);

    const unsigned short* kbase =
        K + ((size_t)(b * SEQ + wave * 256 + ln)) * D_MODEL + h * D_K + 16 * g;
    const int* mbase = mask + ((size_t)b * SEQ + q0) * SEQ + wave * 256 + ln;

#define SC_LOAD(b0_, b1_, ct_)                                                \
    do {                                                                      \
        const unsigned short* kr_ = kbase + (size_t)(ct_) * 16 * D_MODEL;     \
        (b0_).u4[0] = *(const uint4*)(kr_);                                   \
        (b0_).u4[1] = *(const uint4*)(kr_ + 8);                               \
        (b1_).u4[0] = *(const uint4*)(kr_ + 32);                              \
        (b1_).u4[1] = *(const uint4*)(kr_ + 40);                              \
    } while (0)

    auto storeTile = [&](int ct, v8f acc) {
        const int kcol = wave * 256 + ct * 16 + ln;
        for (int r = 0; r < 8; ++r) {
            int qr = r + 8 * g;
            float v = acc[r] * 0.125f;                 // 1/sqrt(64)
            sc[qr * SEQ + kcol] =
                (mbase[(size_t)qr * SEQ + ct * 16] == 0) ? NEG_BIG : v;
        }
    };

    {
        FragAB b0X, b1X, b0Y, b1Y;
        SC_LOAD(b0X, b1X, 0);
        for (int ct = 0; ct < 16; ct += 2) {
            SC_LOAD(b0Y, b1Y, ct + 1);
            v8f accA = {};
            accA = WMMA_BF16(a0, b0X, accA);
            accA = WMMA_BF16(a1, b1X, accA);
            storeTile(ct, accA);
            if (ct + 2 < 16) SC_LOAD(b0X, b1X, ct + 2);
            v8f accB = {};
            accB = WMMA_BF16(a0, b0Y, accB);
            accB = WMMA_BF16(a1, b1Y, accB);
            storeTile(ct + 1, accB);
        }
    }
#undef SC_LOAD
    __syncthreads();

    // ---- softmax: each wave handles 2 rows, wave32 shuffle reductions ---
    for (int rr = 0; rr < 2; ++rr) {
        const int row = wave * 2 + rr;
        float* srow = sc + row * SEQ;
        float m = -3.0e38f;
        for (int k = lane; k < SEQ; k += 32) m = fmaxf(m, srow[k]);
        for (int off = 16; off; off >>= 1)   m = fmaxf(m, __shfl_xor(m, off, 32));
        float s = 0.0f;
        for (int k = lane; k < SEQ; k += 32) {
            float e = __expf(srow[k] - m);
            srow[k] = e;
            s += e;
        }
        for (int off = 16; off; off >>= 1) s += __shfl_xor(s, off, 32);
        const float inv = 1.0f / s;

        float* arow = attnw +
            (((size_t)(b * NUM_HEADS + h) * SEQ) + q0 + row) * SEQ;
        unsigned short* prow = pa + row * SEQ;
        for (int k = lane; k < SEQ; k += 32) {
            float p = srow[k] * inv;
            arow[k] = p;                        // fp32 attn_weights -> HBM
            prow[k] = f32_to_bf16_rne(p);       // bf16 copy stays in LDS
        }
    }
    __syncthreads();

    // ---- out = P @ V : all 8 waves; K range split across wave pairs -----
    // wave & 3 -> 16-wide d slice;  wave >> 2 -> K half [0,1024) / [1024,2048)
    {
        const int nt = wave & 3;
        const int kh = wave >> 2;
        const unsigned short* arow = pa + ln * SEQ + 16 * g;  // A from LDS
        const unsigned short* vcol = Vt + (size_t)b * (D_MODEL * SEQ) +
            (size_t)(h * D_K + nt * 16 + ln) * SEQ + 16 * g;  // B col in V^T

#define PV_LOAD(a_, b_, k_)                                                   \
    do {                                                                      \
        (a_).u4[0] = *(const uint4*)(arow + (k_));                            \
        (a_).u4[1] = *(const uint4*)(arow + (k_) + 8);                        \
        (b_).u4[0] = *(const uint4*)(vcol + (k_));                            \
        (b_).u4[1] = *(const uint4*)(vcol + (k_) + 8);                        \
    } while (0)

        v8f acc = {};
        const int kbeg = kh * (SEQ / 2), kend = kbeg + (SEQ / 2);
        FragAB aX, bX, aY, bY;
        PV_LOAD(aX, bX, kbeg);
        for (int k0 = kbeg; k0 < kend; k0 += 64) {
            PV_LOAD(aY, bY, k0 + 32);
            acc = WMMA_BF16(aX, bX, acc);
            if (k0 + 64 < kend) PV_LOAD(aX, bX, k0 + 64);
            acc = WMMA_BF16(aY, bY, acc);
        }
#undef PV_LOAD

        // combine the two K-halves through LDS (reuse dead fp32 score buffer)
        float* red = sc;                              // [4][16][16] = 4 KB
        if (kh == 1) {
            for (int r = 0; r < 8; ++r)
                red[(nt * 16 + r + 8 * g) * 16 + ln] = acc[r];
        }
        __syncthreads();
        if (kh == 0) {
            for (int r = 0; r < 8; ++r) {
                int q = q0 + r + 8 * g;
                float v = acc[r] + red[(nt * 16 + r + 8 * g) * 16 + ln];
                out[((size_t)(b * SEQ + q)) * D_MODEL + h * D_K + nt * 16 + ln] = v;
            }
        }
    }
}

// ---------------------------------------------------------------------------
extern "C" void kernel_launch(void* const* d_in, const int* in_sizes, int n_in,
                              void* d_out, int out_size, void* d_ws, size_t ws_size,
                              hipStream_t stream)
{
    const float* q   = (const float*)d_in[0];
    const float* key = (const float*)d_in[1];
    const float* val = (const float*)d_in[2];
    const int*   msk = (const int*)  d_in[3];
    const float* Wq  = (const float*)d_in[4];
    const float* bq  = (const float*)d_in[5];
    const float* Wk  = (const float*)d_in[6];
    const float* bk  = (const float*)d_in[7];
    const float* Wv  = (const float*)d_in[8];
    const float* bv  = (const float*)d_in[9];

    float* out   = (float*)d_out;
    float* attnw = out + (size_t)BATCH * SEQ * D_MODEL;   // (out, attn_weights)

    // Workspace layout (bf16 elements): 6*NX + 3*NW  -> ~54 MB total.
    const size_t NX = (size_t)M_ROWS * D_MODEL;   // 4 Mi elements
    const size_t NW = (size_t)D_MODEL * D_MODEL;  // 1 Mi elements
    unsigned short* ws  = (unsigned short*)d_ws;
    unsigned short* Xq  = ws;
    unsigned short* Xk  = Xq  + NX;
    unsigned short* Xv  = Xk  + NX;
    unsigned short* Wqb = Xv  + NX;
    unsigned short* Wkb = Wqb + NW;
    unsigned short* Wvb = Wkb + NW;
    unsigned short* Qb  = Wvb + NW;
    unsigned short* Kb  = Qb  + NX;
    unsigned short* Vtb = Kb  + NX;

    // Stage 0: convert
    cvt_f32_to_bf16<<<4096, 256, 0, stream>>>(q,   Xq,  (int)NX);
    cvt_f32_to_bf16<<<4096, 256, 0, stream>>>(key, Xk,  (int)NX);
    cvt_f32_to_bf16<<<4096, 256, 0, stream>>>(val, Xv,  (int)NX);
    cvt_f32_to_bf16<<<1024, 256, 0, stream>>>(Wq,  Wqb, (int)NW);
    cvt_f32_to_bf16<<<1024, 256, 0, stream>>>(Wk,  Wkb, (int)NW);
    cvt_f32_to_bf16<<<1024, 256, 0, stream>>>(Wv,  Wvb, (int)NW);

    // Stage 1: projections (4096 16x64 wave-tiles / 8 waves per block)
    proj_gemm_bf16<<<512, 256, 0, stream>>>(Xq, Wqb, bq, Qb,  0);
    proj_gemm_bf16<<<512, 256, 0, stream>>>(Xk, Wkb, bk, Kb,  0);
    proj_gemm_bf16<<<512, 256, 0, stream>>>(Xv, Wvb, bv, Vtb, 1);

    // Stage 2: fused scores + softmax + P@V  (B*H*(S/16) = 4096 blocks)
    const size_t shmem = (size_t)16 * SEQ * 4 + (size_t)16 * SEQ * 2; // 192 KB
    attn_fused<<<4096, 256, shmem, stream>>>(Qb, Kb, Vtb, msk, out, attnw);
}
